// MultiHeadAttention_29875792511086
// MI455X (gfx1250) — compile-verified, matrix operand order checked by reading, not run on hardware
//
#include <hip/hip_runtime.h>
#include <hip/hip_bf16.h>
#include <stdint.h>

#define EMBED 768
#define QKV3  2304
#define SEQ   2048
#define BATCH 4
#define HEADS 12
#define HDIM  64

typedef __attribute__((ext_vector_type(16))) __bf16          v16bf;
typedef __attribute__((ext_vector_type(16))) unsigned short  v16u;
typedef __attribute__((ext_vector_type(8)))  float           v8f;
typedef __attribute__((ext_vector_type(4)))  unsigned int    v4u;

__device__ __forceinline__ unsigned short f2bf(float f) {
  unsigned int u = __float_as_uint(f);
  u += 0x7fffu + ((u >> 16) & 1u);          // round-to-nearest-even
  return (unsigned short)(u >> 16);
}

__device__ __forceinline__ v8f wmma_bf16(v16bf a, v16bf b, v8f c) {
  return __builtin_amdgcn_wmma_f32_16x16x32_bf16(
      /*neg_a=*/false, a, /*neg_b=*/false, b,
      /*c_mod=*/(short)0, c, /*reuse_a=*/false, /*reuse_b=*/false);
}

// Fragment loader for contraction-contiguous operands (A, or B of an NT GEMM):
// frag[e] = p[row*ld + K(e)], K(e) = 8*hi + (e&7) + 16*(e>>3)
// -> per lane exactly two contiguous 16-byte runs (b128 loads).
__device__ __forceinline__ v16bf load_frag_nt(const unsigned short* p, int ld) {
  const int lane = threadIdx.x & 31;
  const int row = lane & 15, hi = lane >> 4;
  union { v4u q[2]; v16bf bf; } f;
  const unsigned short* q = p + (size_t)row * ld + 8 * hi;
  f.q[0] = *(const v4u*)(q);
  f.q[1] = *(const v4u*)(q + 16);
  return f.bf;
}

// fp32 -> bf16 elementwise conversion (for x)
__global__ void cvt_bf16(const float* __restrict__ in,
                         unsigned short* __restrict__ out, int nelem) {
  int i = blockIdx.x * blockDim.x + threadIdx.x;
  if (i < nelem) out[i] = f2bf(in[i]);
}

// fp32 [R][C] -> bf16 [C][R] (LDS-tiled 32x32 transpose). grid (C/32, R/32).
__global__ __launch_bounds__(256)
void cvt_transpose_bf16(const float* __restrict__ in,
                        unsigned short* __restrict__ out, int R, int C) {
  __shared__ unsigned short tile[32][33];
  const int r0 = blockIdx.y * 32;
  const int c0 = blockIdx.x * 32;
  const int tr = threadIdx.x >> 5;   // 0..7
  const int tc = threadIdx.x & 31;
#pragma unroll
  for (int i = 0; i < 4; ++i)
    tile[tr + 8 * i][tc] = f2bf(in[(size_t)(r0 + tr + 8 * i) * C + c0 + tc]);
  __syncthreads();
#pragma unroll
  for (int i = 0; i < 4; ++i)
    out[(size_t)(c0 + tr + 8 * i) * R + r0 + tc] = tile[tc][tr + 8 * i];
}

// bf16 qkv V-part [b][key][h*64+d] -> Vt [b][h][d][key]. grid (SEQ/32, H, B).
__global__ __launch_bounds__(256)
void transpose_v(const unsigned short* __restrict__ qkv,
                 unsigned short* __restrict__ vt) {
  __shared__ __align__(16) unsigned short tile[32][72];  // 32 keys x 64 d (+pad)
  const int b = blockIdx.z, h = blockIdx.y;
  const int k0 = blockIdx.x * 32;
  const int t = threadIdx.x;
  {
    const int key = t >> 3, d8 = (t & 7) << 3;           // 8 contiguous d per thread
    const unsigned short* src = qkv + (size_t)(b * SEQ + k0 + key) * QKV3
                                + 2 * EMBED + h * HDIM + d8;
    *(v4u*)&tile[key][d8] = *(const v4u*)src;
  }
  __syncthreads();
  {
    const int d = t >> 2, key8 = (t & 3) << 3;           // 8 contiguous keys out
    union { v4u q; unsigned short s[8]; } u;
#pragma unroll
    for (int j = 0; j < 8; ++j) u.s[j] = tile[key8 + j][d];
    *(v4u*)(vt + ((size_t)(b * HEADS + h) * HDIM + d) * SEQ + k0 + key8) = u.q;
  }
}

// C[M,N] = A[M,K] * Bt[N,K]^T + bias. A,Bt bf16 contraction-contiguous.
// One wave computes a 32x64 tile (8 WMMA C-tiles), K stepped by 32.
__global__ __launch_bounds__(256)
void wmma_gemm(const unsigned short* __restrict__ A,
               const unsigned short* __restrict__ Bt,
               const float* __restrict__ bias,
               unsigned short* __restrict__ Cb, float* __restrict__ Cf,
               int M, int K, int N) {
  const int wave = threadIdx.x >> 5;
  const int tile = blockIdx.x * (blockDim.x >> 5) + wave;
  const int ntiles = N >> 6;
  const int mt = tile / ntiles;
  const int nt = tile % ntiles;
  if (mt * 32 >= M) return;

  v8f acc[2][4] = {};
  const unsigned short* A0 = A + (size_t)mt * 32 * K;
  const unsigned short* B0 = Bt + (size_t)nt * 64 * K;
  for (int k0 = 0; k0 < K; k0 += 32) {
    const v16bf a0 = load_frag_nt(A0 + k0, K);
    const v16bf a1 = load_frag_nt(A0 + (size_t)16 * K + k0, K);
#pragma unroll
    for (int c = 0; c < 4; ++c) {
      const v16bf bf = load_frag_nt(B0 + (size_t)(c * 16) * K + k0, K);
      acc[0][c] = wmma_bf16(a0, bf, acc[0][c]);
      acc[1][c] = wmma_bf16(a1, bf, acc[1][c]);
    }
  }

  const int lane = threadIdx.x & 31;
  const int n = lane & 15, hi = lane >> 4;
#pragma unroll
  for (int mi = 0; mi < 2; ++mi)
#pragma unroll
    for (int c = 0; c < 4; ++c) {
      const int col = nt * 64 + c * 16 + n;
      const float bv = bias ? bias[col] : 0.f;
#pragma unroll
      for (int r = 0; r < 8; ++r) {
        const int row = mt * 32 + mi * 16 + r + 8 * hi;
        const float v = acc[mi][c][r] + bv;
        if (Cb) Cb[(size_t)row * N + col] = f2bf(v);
        else    Cf[(size_t)row * N + col] = v;
      }
    }
}

// Flash attention: grid (SEQ/64, HEADS, BATCH), 128 threads (4 waves),
// each wave owns one 16-query tile and streams keys in blocks of 32.
__global__ __launch_bounds__(128)
void flash_attn(const unsigned short* __restrict__ qkv,
                const unsigned short* __restrict__ vt,
                unsigned short* __restrict__ out) {
  __shared__ __align__(16) unsigned short p_lds[4][16 * 32];
  const int wave = threadIdx.x >> 5;
  const int lane = threadIdx.x & 31;
  const int n = lane & 15, hi = lane >> 4;
  const int b = blockIdx.z, h = blockIdx.y;
  const int q0 = (blockIdx.x * 4 + wave) * 16;
  const size_t base = (size_t)b * SEQ * QKV3;
  const unsigned short* Qp = qkv + base + (size_t)q0 * QKV3 + h * HDIM;
  const unsigned short* Kp = qkv + base + EMBED + h * HDIM;          // ld=QKV3
  const unsigned short* Vtp = vt + (size_t)(b * HEADS + h) * HDIM * SEQ;  // [d][key]
  unsigned short* pl = p_lds[wave];

  const v16bf qa0 = load_frag_nt(Qp, QKV3);        // d 0..31
  const v16bf qa1 = load_frag_nt(Qp + 32, QKV3);   // d 32..63

  v8f o[4] = {};
  float mrow[8], lrow[8];
#pragma unroll
  for (int r = 0; r < 8; ++r) { mrow[r] = -1e30f; lrow[r] = 0.f; }

  for (int j0 = 0; j0 < SEQ; j0 += 32) {
    // S = scale * Q * K^T (two 16x16 tiles over this 32-key block)
    float sv[2][8];
#pragma unroll
    for (int t = 0; t < 2; ++t) {
      const unsigned short* Kt = Kp + (size_t)(j0 + 16 * t) * QKV3;
      v8f s = {};
      s = wmma_bf16(qa0, load_frag_nt(Kt, QKV3), s);
      s = wmma_bf16(qa1, load_frag_nt(Kt + 32, QKV3), s);
#pragma unroll
      for (int r = 0; r < 8; ++r) sv[t][r] = s[r] * 0.125f;  // 1/sqrt(64)
    }
    // per-row block max across the 16 key-columns (lanes of this half)
    float bm[8];
#pragma unroll
    for (int r = 0; r < 8; ++r) bm[r] = fmaxf(sv[0][r], sv[1][r]);
#pragma unroll
    for (int m = 8; m >= 1; m >>= 1)
#pragma unroll
      for (int r = 0; r < 8; ++r) bm[r] = fmaxf(bm[r], __shfl_xor(bm[r], m, 32));
    // online-softmax rescale
    float alpha[8];
#pragma unroll
    for (int r = 0; r < 8; ++r) {
      const float mn = fmaxf(mrow[r], bm[r]);
      alpha[r] = __expf(mrow[r] - mn);
      mrow[r] = mn;
      lrow[r] *= alpha[r];
    }
#pragma unroll
    for (int c = 0; c < 4; ++c)
#pragma unroll
      for (int r = 0; r < 8; ++r) o[c][r] *= alpha[r];
    // P = exp(S - m) -> 16x32 bf16 tile in LDS; accumulate row sums
    float rs[8];
#pragma unroll
    for (int r = 0; r < 8; ++r) rs[r] = 0.f;
#pragma unroll
    for (int t = 0; t < 2; ++t)
#pragma unroll
      for (int r = 0; r < 8; ++r) {
        const float p = __expf(sv[t][r] - mrow[r]);
        rs[r] += p;
        pl[(r + 8 * hi) * 32 + t * 16 + n] = f2bf(p);
      }
#pragma unroll
    for (int m = 8; m >= 1; m >>= 1)
#pragma unroll
      for (int r = 0; r < 8; ++r) rs[r] += __shfl_xor(rs[r], m, 32);
#pragma unroll
    for (int r = 0; r < 8; ++r) lrow[r] += rs[r];
    // O += P(16x32) * V(32x64); V fragments from Vt are contiguous b128 loads
    const v16bf pa = load_frag_nt(pl, 32);
#pragma unroll
    for (int c = 0; c < 4; ++c)
      o[c] = wmma_bf16(pa, load_frag_nt(Vtp + (size_t)(c * 16) * SEQ + j0, SEQ),
                       o[c]);
  }

#pragma unroll
  for (int c = 0; c < 4; ++c)
#pragma unroll
    for (int r = 0; r < 8; ++r) {
      const float v = o[c][r] / lrow[r];
      out[((size_t)b * SEQ + q0 + r + 8 * hi) * EMBED + h * HDIM + c * 16 + n] =
          f2bf(v);
    }
}

extern "C" void kernel_launch(void* const* d_in, const int* in_sizes, int n_in,
                              void* d_out, int out_size, void* d_ws, size_t ws_size,
                              hipStream_t stream) {
  const float* x      = (const float*)d_in[0];
  const float* W_qkv  = (const float*)d_in[1];
  const float* b_qkv  = (const float*)d_in[2];
  const float* W_proj = (const float*)d_in[3];
  const float* b_proj = (const float*)d_in[4];

  // workspace layout (bf16 / ushort elements), ~80 MB total
  unsigned short* ws  = (unsigned short*)d_ws;
  unsigned short* xb  = ws;                                  // [8192, 768]
  unsigned short* wqt = xb + (size_t)8192 * EMBED;           // [2304, 768] = W_qkv^T
  unsigned short* wpt = wqt + (size_t)QKV3 * EMBED;          // [768, 768]  = W_proj^T
  unsigned short* qk  = wpt + (size_t)EMBED * EMBED;         // [8192, 2304]
  unsigned short* ao  = qk + (size_t)8192 * QKV3;            // [8192, 768]
  unsigned short* vtb = ao + (size_t)8192 * EMBED;           // [4,12,64,2048]

  const int M = BATCH * SEQ;  // 8192

  cvt_bf16<<<(M * EMBED + 255) / 256, 256, 0, stream>>>(x, xb, M * EMBED);
  cvt_transpose_bf16<<<dim3(QKV3 / 32, EMBED / 32), 256, 0, stream>>>(
      W_qkv, wqt, EMBED, QKV3);
  cvt_transpose_bf16<<<dim3(EMBED / 32, EMBED / 32), 256, 0, stream>>>(
      W_proj, wpt, EMBED, EMBED);

  // QKV projection: [8192,768] x [768,2304] + b_qkv -> bf16 qkv
  {
    const int tiles = (M / 32) * (QKV3 / 64);  // 256*36 = 9216
    wmma_gemm<<<tiles / 8, 256, 0, stream>>>(xb, wqt, b_qkv, qk, nullptr,
                                             M, EMBED, QKV3);
  }

  // V transpose to [b][h][d][key] for contiguous PV fragments
  transpose_v<<<dim3(SEQ / 32, HEADS, BATCH), 256, 0, stream>>>(qk, vtb);

  // fused flash attention -> bf16 [8192, 768]
  flash_attn<<<dim3(SEQ / 64, HEADS, BATCH), 128, 0, stream>>>(qk, vtb, ao);

  // output projection: [8192,768] x [768,768] + b_proj -> fp32 d_out
  {
    const int tiles = (M / 32) * (EMBED / 64);  // 256*12 = 3072
    wmma_gemm<<<tiles / 8, 256, 0, stream>>>(ao, wpt, b_proj, nullptr,
                                             (float*)d_out, M, EMBED, EMBED);
  }
}
